// WordCharEncoderBiRNN_73976516706550
// MI455X (gfx1250) — compile-verified, matrix operand order checked by reading, not run on hardware
//
#include <hip/hip_runtime.h>
#include <hip/hip_bf16.h>
#include <math.h>

#define S_LEN  4096
#define CHAR_L 16
#define F_DIM  256
#define H_DIM  512
#define G_DIM  2048   // 4*H

typedef __attribute__((ext_vector_type(16))) __bf16 v16bf;
typedef __attribute__((ext_vector_type(2)))  __bf16 v2bf;
typedef __attribute__((ext_vector_type(8)))  float  v8f;

__device__ __forceinline__ __bf16 f2bf(float f) {
  union { float f; unsigned u; } v; v.f = f;
  unsigned u = v.u;
  u += ((u >> 16) & 1u) + 0x7fffu;                 // round-to-nearest-even
  union { unsigned short s; __bf16 b; } o; o.s = (unsigned short)(u >> 16);
  return o.b;
}
__device__ __forceinline__ float bflo(unsigned u) {
  union { unsigned u; float f; } v; v.u = u << 16; return v.f;
}
__device__ __forceinline__ float bfhi(unsigned u) {
  union { unsigned u; float f; } v; v.u = u & 0xffff0000u; return v.f;
}
__device__ __forceinline__ float sigmoidf(float x) { return 1.0f / (1.0f + __expf(-x)); }

// ---------------- small utility kernels ----------------
__global__ void cvt_f32_bf16(const float* __restrict__ x, __bf16* __restrict__ y, int n) {
  int i = blockIdx.x * blockDim.x + threadIdx.x;
  if (i < n) y[i] = f2bf(x[i]);
}
__global__ void zero_f32(float* __restrict__ p, int n) {
  int i = blockIdx.x * blockDim.x + threadIdx.x;
  if (i < n) p[i] = 0.0f;
}
__global__ void bias_sum(const float* __restrict__ a, const float* __restrict__ b,
                         float* __restrict__ o, int n) {
  int i = blockIdx.x * blockDim.x + threadIdx.x;
  if (i < n) o[i] = a[i] + b[i];
}
__global__ void gather_embed_bf16(const int* __restrict__ idx, const float* __restrict__ emb,
                                  __bf16* __restrict__ ce, int SL, int F) {
  int i = blockIdx.x * blockDim.x + threadIdx.x;
  if (i >= SL * F) return;
  int sl = i / F, f = i - sl * F;
  ce[i] = f2bf(emb[idx[sl] * F + f]);
}
__global__ void build_x_bf16(const float* __restrict__ word, const float* __restrict__ cf,
                             __bf16* __restrict__ x) {
  int i = blockIdx.x * blockDim.x + threadIdx.x;   // S*512
  if (i >= S_LEN * 512) return;
  int s = i >> 9, j = i & 511;
  float v = (j < 256) ? word[s * 256 + j] : cf[s * 256 + (j - 256)];
  x[i] = f2bf(v);
}

// ---- WMMA GEMM: C[M,N] = A[M,K] * Bt[N,K]^T (+bias); 16Mx64N tile per wave ----
// 128-thread block = 4 waves, wave w owns M-tile (blockIdx.y*4+w) and the four
// N-tiles blockIdx.x*4 .. +3, so each A fragment feeds 4 WMMAs per K-step.
// A row r addressed as A + (r/rpo)*strideOuter + (r%rpo)*strideInner so the
// conv einsum reads its [ce(s,l), ce(s,l+1)] windows directly from the
// [S,16,256] embed buffer.
// EPI==0: store f32.  EPI==1: fused relu+maxpool over 15 rows via atomicMax
// (valid since relu output >= 0 -> int compare == float compare).
template <int EPI>
__global__ void __launch_bounds__(128)
gemm_wmma_bf16(const __bf16* __restrict__ A,
               const __bf16* __restrict__ Bt,
               const float* __restrict__ bias,
               float* __restrict__ C,
               int* __restrict__ pool,
               int M, int N, int K,
               int rpo, int strideOuter, int strideInner,
               int reverseA) {
  const int lane = threadIdx.x & 31;
  const int wid  = threadIdx.x >> 5;
  const int m  = lane & 15;
  const int hi = lane >> 4;
  const int tn0 = blockIdx.x * 4;
  const int tm  = blockIdx.y * 4 + wid;

  int r = tm * 16 + m;
  if (reverseA) r = M - 1 - r;
  const __bf16* arow  = A + (size_t)(r / rpo) * strideOuter + (size_t)(r % rpo) * strideInner;
  const __bf16* brow0 = Bt + (size_t)(tn0 * 16 + m) * K;   // +t*16*K per sub-tile
  const size_t  bstep = (size_t)16 * K;

  v8f acc0 = {}, acc1 = {}, acc2 = {}, acc3 = {};
  #pragma unroll 2
  for (int k0 = 0; k0 < K; k0 += 32) {
    v16bf a;
    #pragma unroll
    for (int j = 0; j < 4; ++j) {
      int ka = k0 + hi * 8 + 2 * j;       // low-half K group for this half-wave
      int kb = ka + 16;                   // high-half K group
      v2bf a0 = *(const v2bf*)(arow + ka);
      v2bf a1 = *(const v2bf*)(arow + kb);
      a[2*j] = a0[0]; a[2*j+1] = a0[1];
      a[8+2*j] = a1[0]; a[8+2*j+1] = a1[1];
    }
    v16bf b0, b1, b2, b3;
    #pragma unroll
    for (int j = 0; j < 4; ++j) {
      int ka = k0 + hi * 8 + 2 * j;
      int kb = ka + 16;
      v2bf p, q;
      p = *(const v2bf*)(brow0 + 0*bstep + ka); q = *(const v2bf*)(brow0 + 0*bstep + kb);
      b0[2*j] = p[0]; b0[2*j+1] = p[1]; b0[8+2*j] = q[0]; b0[8+2*j+1] = q[1];
      p = *(const v2bf*)(brow0 + 1*bstep + ka); q = *(const v2bf*)(brow0 + 1*bstep + kb);
      b1[2*j] = p[0]; b1[2*j+1] = p[1]; b1[8+2*j] = q[0]; b1[8+2*j+1] = q[1];
      p = *(const v2bf*)(brow0 + 2*bstep + ka); q = *(const v2bf*)(brow0 + 2*bstep + kb);
      b2[2*j] = p[0]; b2[2*j+1] = p[1]; b2[8+2*j] = q[0]; b2[8+2*j+1] = q[1];
      p = *(const v2bf*)(brow0 + 3*bstep + ka); q = *(const v2bf*)(brow0 + 3*bstep + kb);
      b3[2*j] = p[0]; b3[2*j+1] = p[1]; b3[8+2*j] = q[0]; b3[8+2*j+1] = q[1];
    }
    acc0 = __builtin_amdgcn_wmma_f32_16x16x32_bf16(false, a, false, b0, (short)0, acc0, false, false);
    acc1 = __builtin_amdgcn_wmma_f32_16x16x32_bf16(false, a, false, b1, (short)0, acc1, false, false);
    acc2 = __builtin_amdgcn_wmma_f32_16x16x32_bf16(false, a, false, b2, (short)0, acc2, false, false);
    acc3 = __builtin_amdgcn_wmma_f32_16x16x32_bf16(false, a, false, b3, (short)0, acc3, false, false);
  }

  #pragma unroll
  for (int t = 0; t < 4; ++t) {
    const v8f acc = (t == 0) ? acc0 : (t == 1) ? acc1 : (t == 2) ? acc2 : acc3;
    const int cn  = (tn0 + t) * 16 + (lane & 15);
    const float bv = bias ? bias[cn] : 0.0f;
    #pragma unroll
    for (int i = 0; i < 8; ++i) {
      int cm = tm * 16 + hi * 8 + i;
      float v = acc[i] + bv;
      if (EPI == 0) {
        C[(size_t)cm * N + cn] = v;
      } else {
        int s = cm / 15;                       // 15 conv rows per sequence pos
        v = fmaxf(v, 0.0f);                    // relu folded into max(0,·)
        atomicMax(pool + (size_t)s * N + cn, __float_as_int(v));
      }
    }
  }
}

// ---------------- persistent bidirectional LSTM recurrence ----------------
// One 32-wave block per direction; h,c in LDS; Whh streamed as bf16 from L2.
__global__ void __launch_bounds__(1024)
lstm_recurrence(const float* __restrict__ gates_all, const __bf16* __restrict__ Whh_all,
                float* __restrict__ out, float* __restrict__ hidden,
                float* __restrict__ cell) {
  const int dir = blockIdx.x;
  const int tid = threadIdx.x;
  const float*  gin = gates_all + (size_t)dir * S_LEN * G_DIM;
  const __bf16* W   = Whh_all  + (size_t)dir * G_DIM * H_DIM;

  __shared__ float hs[H_DIM];
  __shared__ float cs[H_DIM];
  __shared__ float gb[G_DIM];

  if (tid < H_DIM) { hs[tid] = 0.0f; cs[tid] = 0.0f; }
  __syncthreads();

  for (int t = 0; t < S_LEN; ++t) {
    const float* gt = gin + (size_t)t * G_DIM;
    if (t + 1 < S_LEN)
      __builtin_prefetch(gin + (size_t)(t + 1) * G_DIM + tid * 2, 0, 1);
    #pragma unroll
    for (int half = 0; half < 2; ++half) {
      int j = tid + half * 1024;
      const uint4* w4 = (const uint4*)(W + (size_t)j * H_DIM);
      float acc = gt[j];
      #pragma unroll 8
      for (int q = 0; q < H_DIM / 8; ++q) {
        uint4 w = w4[q];
        int k = q * 8;
        acc += bflo(w.x) * hs[k+0] + bfhi(w.x) * hs[k+1]
             + bflo(w.y) * hs[k+2] + bfhi(w.y) * hs[k+3]
             + bflo(w.z) * hs[k+4] + bfhi(w.z) * hs[k+5]
             + bflo(w.w) * hs[k+6] + bfhi(w.w) * hs[k+7];
      }
      gb[j] = acc;
    }
    __syncthreads();
    if (tid < H_DIM) {
      float ig = sigmoidf(gb[tid]);
      float fg = sigmoidf(gb[H_DIM + tid]);
      float gg = tanhf(gb[2 * H_DIM + tid]);
      float og = sigmoidf(gb[3 * H_DIM + tid]);
      float c = fg * cs[tid] + ig * gg;
      float h = og * tanhf(c);
      cs[tid] = c; hs[tid] = h;
      int pos = dir ? (S_LEN - 1 - t) : t;
      out[(size_t)pos * (2 * H_DIM) + dir * H_DIM + tid] = h;
      if (t == S_LEN - 1) {
        hidden[dir * H_DIM + tid] = h;
        cell[dir * H_DIM + tid]   = c;
      }
    }
    __syncthreads();
  }
}

// ---------------- host launch ----------------
extern "C" void kernel_launch(void* const* d_in, const int* in_sizes, int n_in,
                              void* d_out, int out_size, void* d_ws, size_t ws_size,
                              hipStream_t stream) {
  (void)in_sizes; (void)n_in; (void)out_size; (void)ws_size;
  const float* word   = (const float*)d_in[0];
  const int*   cidx   = (const int*)  d_in[1];
  const float* cemb   = (const float*)d_in[2];
  const float* conv_w = (const float*)d_in[3];
  const float* conv_b = (const float*)d_in[4];
  const float* lin_w  = (const float*)d_in[5];
  const float* lin_b  = (const float*)d_in[6];
  const float* Wih_f  = (const float*)d_in[7];
  const float* Whh_f  = (const float*)d_in[8];
  const float* bih_f  = (const float*)d_in[9];
  const float* bhh_f  = (const float*)d_in[10];
  const float* Wih_r  = (const float*)d_in[11];
  const float* Whh_r  = (const float*)d_in[12];
  const float* bih_r  = (const float*)d_in[13];
  const float* bhh_r  = (const float*)d_in[14];

  char* ws = (char*)d_ws;
  size_t off = 0;
  auto wsalloc = [&](size_t bytes) -> void* {
    void* p = ws + off; off += (bytes + 255) & ~(size_t)255; return p;
  };

  float*  gates  = (float*) wsalloc((size_t)2 * S_LEN * G_DIM * 4);  // 67 MB
  __bf16* ce     = (__bf16*)gates;            // 33.5 MB overlay; dead before gates written
  float*  poolf  = (float*) wsalloc((size_t)S_LEN * F_DIM * 4);
  __bf16* poolb  = (__bf16*)wsalloc((size_t)S_LEN * F_DIM * 2);
  __bf16* convWb = (__bf16*)wsalloc((size_t)F_DIM * 512 * 2);
  __bf16* linWb  = (__bf16*)wsalloc((size_t)F_DIM * F_DIM * 2);
  float*  charf  = (float*) wsalloc((size_t)S_LEN * F_DIM * 4);
  __bf16* xb     = (__bf16*)wsalloc((size_t)S_LEN * H_DIM * 2);
  __bf16* WihB   = (__bf16*)wsalloc((size_t)2 * G_DIM * H_DIM * 2);
  __bf16* WhhB   = (__bf16*)wsalloc((size_t)2 * G_DIM * H_DIM * 2);
  float*  bsum   = (float*) wsalloc((size_t)2 * G_DIM * 4);

  float* out_seq  = (float*)d_out;
  float* out_hid  = out_seq + (size_t)S_LEN * 2 * H_DIM;
  float* out_cell = out_hid + 2 * H_DIM;

  const int TB = 256;
  auto nb = [&](long n) { return dim3((unsigned)((n + TB - 1) / TB)); };

  // weight conversions + bias sums + char-embed gather
  cvt_f32_bf16<<<nb(F_DIM*512),   TB, 0, stream>>>(conv_w, convWb, F_DIM*512);
  cvt_f32_bf16<<<nb(F_DIM*F_DIM), TB, 0, stream>>>(lin_w,  linWb,  F_DIM*F_DIM);
  cvt_f32_bf16<<<nb(G_DIM*H_DIM), TB, 0, stream>>>(Wih_f, WihB,                       G_DIM*H_DIM);
  cvt_f32_bf16<<<nb(G_DIM*H_DIM), TB, 0, stream>>>(Wih_r, WihB + (size_t)G_DIM*H_DIM, G_DIM*H_DIM);
  cvt_f32_bf16<<<nb(G_DIM*H_DIM), TB, 0, stream>>>(Whh_f, WhhB,                       G_DIM*H_DIM);
  cvt_f32_bf16<<<nb(G_DIM*H_DIM), TB, 0, stream>>>(Whh_r, WhhB + (size_t)G_DIM*H_DIM, G_DIM*H_DIM);
  bias_sum<<<nb(G_DIM), TB, 0, stream>>>(bih_f, bhh_f, bsum,         G_DIM);
  bias_sum<<<nb(G_DIM), TB, 0, stream>>>(bih_r, bhh_r, bsum + G_DIM, G_DIM);
  gather_embed_bf16<<<nb((long)S_LEN*CHAR_L*F_DIM), TB, 0, stream>>>(cidx, cemb, ce,
                                                                     S_LEN*CHAR_L, F_DIM);
  zero_f32<<<nb(S_LEN*F_DIM), TB, 0, stream>>>(poolf, S_LEN*F_DIM);

  // conv einsum as WMMA GEMM, fused relu+maxpool: M=61440, N=256, K=512
  {
    dim3 g(F_DIM/64, (S_LEN*15)/16/4);
    gemm_wmma_bf16<1><<<g, 128, 0, stream>>>(ce, convWb, conv_b, nullptr, (int*)poolf,
        S_LEN*15, F_DIM, 512, /*rpo=*/15, /*sOuter=*/CHAR_L*F_DIM, /*sInner=*/F_DIM, 0);
  }
  cvt_f32_bf16<<<nb(S_LEN*F_DIM), TB, 0, stream>>>(poolf, poolb, S_LEN*F_DIM);

  // linear: charf = pooled @ lin_w^T + lin_b
  {
    dim3 g(F_DIM/64, S_LEN/16/4);
    gemm_wmma_bf16<0><<<g, 128, 0, stream>>>(poolb, linWb, lin_b, charf, nullptr,
        S_LEN, F_DIM, F_DIM, 1, F_DIM, 0, 0);
  }
  build_x_bf16<<<nb((long)S_LEN*H_DIM), TB, 0, stream>>>(word, charf, xb);

  // gate-input GEMMs (reverse direction reads x time-reversed)
  {
    dim3 g(G_DIM/64, S_LEN/16/4);
    gemm_wmma_bf16<0><<<g, 128, 0, stream>>>(xb, WihB, bsum, gates, nullptr,
        S_LEN, G_DIM, H_DIM, 1, H_DIM, 0, 0);
    gemm_wmma_bf16<0><<<g, 128, 0, stream>>>(xb, WihB + (size_t)G_DIM*H_DIM, bsum + G_DIM,
        gates + (size_t)S_LEN*G_DIM, nullptr,
        S_LEN, G_DIM, H_DIM, 1, H_DIM, 0, 1);
  }

  // sequential recurrence, both directions concurrently
  lstm_recurrence<<<2, 1024, 0, stream>>>(gates, WhhB, out_seq, out_hid, out_cell);
}
